// LSTM_Unit_63376537420221
// MI455X (gfx1250) — matrix operand order repaired and checked by reading, not compiled, not run on hardware
//
#include <hip/hip_runtime.h>
#include <hip/hip_bf16.h>
#include <math.h>

#define BS   16384
#define XRW  768
#define HD   256
#define DIN  1280
#define MT   128           // M tile per block
#define HT   32            // h-slice per block
#define NC   128           // 4 gates * HT columns in C tile
#define KS   32            // K step (bf16 wmma K)
#define NKT  (DIN / KS)    // 40
#define BTILE_BYTES  (NC * KS * 2)         // 8192 B per B tile
#define BTILE_E8     (BTILE_BYTES / 8)     // 1024 8-byte elements
#define WS_NEEDED ((size_t)(HD / HT) * NKT * BTILE_BYTES)   // 2.62 MB

typedef __bf16 bf16;
typedef __attribute__((ext_vector_type(4)))  float        f32x4;
typedef __attribute__((ext_vector_type(8)))  float        f32x8;
typedef __attribute__((ext_vector_type(4)))  bf16         bf16x4;
typedef __attribute__((ext_vector_type(8)))  bf16         bf16x8;
typedef __attribute__((ext_vector_type(16))) bf16         bf16x16;
typedef __attribute__((ext_vector_type(4)))  unsigned int u32x4;
typedef __attribute__((ext_vector_type(4)))  int          i32x4;
typedef __attribute__((ext_vector_type(8)))  int          i32x8;

__device__ __forceinline__ float sigmoid_(float x) {
    return 1.0f / (1.0f + __expf(-x));
}
__device__ __forceinline__ float tanh_(float x) {
    float a = __expf(2.0f * x);
    return (a - 1.0f) / (a + 1.0f);
}

// ---------------------------------------------------------------------------
// Weight pre-conversion: fp32 [k][256] x 4 gates  ->  bf16 B-tile images.
// Tile (hs, kt) is 128 cols x 32 k, stored col-major [c][k] contiguously so
// the GEMM kernel can TDM/async-DMA it straight into LDS with no reformatting.
// ---------------------------------------------------------------------------
__global__ __launch_bounds__(256)
void conv_w_kernel(const float* __restrict__ Wi, const float* __restrict__ Wf,
                   const float* __restrict__ Wo, const float* __restrict__ Ws,
                   bf16* __restrict__ out)
{
    const int hs = blockIdx.x;        // 0..7   h-slice
    const int kt = blockIdx.y;        // 0..39  k-tile
    const int t  = threadIdx.x;
    const int c  = t & 127;           // tile column
    const int kh = t >> 7;            // 0/1: k half
    const int gate = c >> 5;
    const float* W = (gate == 0) ? Wi : (gate == 1) ? Wf : (gate == 2) ? Wo : Ws;
    const float* src = W + (size_t)(kt * KS + kh * 16) * HD + hs * HT + (c & 31);
    bf16x8 v0, v1;
#pragma unroll
    for (int k = 0; k < 8; ++k) v0[k] = (bf16)src[(size_t)k * HD];
#pragma unroll
    for (int k = 0; k < 8; ++k) v1[k] = (bf16)src[(size_t)(k + 8) * HD];
    bf16* dst = out + (((size_t)hs * NKT + kt) * NC + c) * KS + kh * 16;
    *(bf16x8*)dst       = v0;
    *(bf16x8*)(dst + 8) = v1;
}

// ---------------------------------------------------------------------------
// Main fused kernel: bf16 WMMA GEMM (A converted in-kernel, B moved by the
// Tensor Data Mover from preconverted workspace) + LSTM gate epilogue.
// LDS map: A buffers @ [0, 16KB), B buffers @ [16KB, 32KB), C tile reuses [0, 64KB).
// ---------------------------------------------------------------------------
__global__ __launch_bounds__(256)
void lstm_wmma_async(const float* __restrict__ x,
                     const float* __restrict__ pt,
                     const float* __restrict__ pl,
                     const float* __restrict__ old_state,
                     const bf16*  __restrict__ wbf,
                     const float* __restrict__ bi, const float* __restrict__ bfv,
                     const float* __restrict__ bo, const float* __restrict__ bs,
                     float* __restrict__ outH, float* __restrict__ outS)
{
    extern __shared__ char smem_raw[];
    bf16*  const Albase = (bf16*)smem_raw;               // 2 x [128 x 32] bf16
    bf16*  const Blbase = (bf16*)(smem_raw + 16384);     // 2 x [128 c x 32 k] bf16
    float* const glds   = (float*)smem_raw;              // [128 x 128] f32 (reused)

    const int tid  = threadIdx.x;
    const int lane = tid & 31;
    const int wave = tid >> 5;
    const int l15  = lane & 15;
    const int kh   = lane >> 4;
    const int wm   = wave & 3;
    const int wn   = wave >> 2;

    const int mBase = blockIdx.y * MT;
    const int hOff  = blockIdx.x * HT;

    // A global-load geometry (const per thread)
    const int ac4 = (tid & 7) * 4;
    const int ar0 = tid >> 3;

    f32x4 areg[4];

    auto ldA = [&](int kt) {
        const int k0 = kt * KS;
        const float* src; int stride;
        if (k0 < XRW)           { src = x  + k0;              stride = XRW; }
        else if (k0 < XRW + HD) { src = pt + (k0 - XRW);      stride = HD;  }
        else                    { src = pl + (k0 - XRW - HD); stride = HD;  }
        src += (size_t)(mBase + ar0) * stride + ac4;
#pragma unroll
        for (int j = 0; j < 4; ++j)
            areg[j] = *(const f32x4*)(src + (size_t)(32 * j) * stride);
    };
    auto stA = [&](int buf) {
        bf16* Ab = Albase + buf * (MT * KS);
#pragma unroll
        for (int j = 0; j < 4; ++j) {
            bf16x4 v;
#pragma unroll
            for (int e = 0; e < 4; ++e) v[e] = (bf16)areg[j][e];
            *(bf16x4*)(Ab + (ar0 + 32 * j) * KS + ac4) = v;   // ds_store_b64
        }
    };

#if __has_builtin(__builtin_amdgcn_tensor_load_to_lds)
    // ---- Tensor Data Mover path: one D# per 8KB B tile, issued by wave 0 ----
    auto issueB = [&](int kt, int buf) {
        if (wave == 0) {
            const unsigned ldsAddr = 16384u + (unsigned)buf * 8192u;
            const unsigned long long ga =
                (unsigned long long)(const void*)wbf +
                (unsigned long long)((unsigned)(blockIdx.x * NKT + kt)) * BTILE_BYTES;
            u32x4 g0 = {};
            g0[0] = 1u;                                        // count=1, user mode
            g0[1] = ldsAddr;                                   // lds_addr (bytes)
            g0[2] = (unsigned)(ga & 0xffffffffull);            // global_addr[31:0]
            g0[3] = (unsigned)((ga >> 32) & 0x01ffffffull)     // global_addr[56:32]
                  | (2u << 30);                                // type = 2 ("image")
            i32x8 g1 = {};
            g1[0] = (int)(3u << 16);                           // data_size=3 (8B), mask=0
            g1[1] = (int)((BTILE_E8 & 0xffffu) << 16);         // tensor_dim0[15:0]
            g1[2] = (int)((BTILE_E8 >> 16) | (1u << 16));      // tensor_dim0[31:16], tensor_dim1=1
            g1[3] = (int)((unsigned)BTILE_E8 << 16);           // tile_dim0 = 1024
            g1[5] = (int)BTILE_E8;                             // tensor_dim0_stride
            i32x4 gz = {};
#if defined(__clang_major__) && (__clang_major__ >= 23)
            i32x8 gz8 = {};
            __builtin_amdgcn_tensor_load_to_lds(g0, g1, gz, gz, gz8, 0);
#else
            __builtin_amdgcn_tensor_load_to_lds(g0, g1, gz, gz, 0);
#endif
        }
    };
    auto waitB = [&]() {
        if (wave == 0) __builtin_amdgcn_s_wait_tensorcnt(0);
    };
#else
    // ---- fallback: per-lane async load to LDS (32B per lane) ----
    const unsigned t32 = (unsigned)tid * 32u;
    auto issueB = [&](int kt, int buf) {
        unsigned gl = (unsigned)((blockIdx.x * NKT + kt) * BTILE_BYTES) + t32;
        unsigned ld = 16384u + (unsigned)buf * 8192u + t32;
        asm volatile("global_load_async_to_lds_b128 %0, %1, %2\n\t"
                     "global_load_async_to_lds_b128 %0, %1, %2 offset:16"
                     :: "v"(ld), "v"(gl), "s"(wbf) : "memory");
    };
    auto waitB = [&]() {
        asm volatile("s_wait_asynccnt 0x0" ::: "memory");
    };
#endif

    f32x8 acc[2][4] = {};

    auto compute = [&](int buf) {
        const bf16* Ab = Albase + buf * (MT * KS);
        const bf16* Bb = Blbase + buf * (KS * NC);
        union U16 { bf16x16 v; bf16x8 h[2]; };
        U16 afrag[2], bfrag[4];
#pragma unroll
        for (int tm = 0; tm < 2; ++tm) {
            const bf16* p = Ab + (wm * 32 + tm * 16 + l15) * KS + kh * 8;
            afrag[tm].h[0] = *(const bf16x8*)p;
            afrag[tm].h[1] = *(const bf16x8*)(p + 16);
        }
#pragma unroll
        for (int tn = 0; tn < 4; ++tn) {
            const bf16* p = Bb + (wn * 64 + tn * 16 + l15) * KS + kh * 8;
            bfrag[tn].h[0] = *(const bf16x8*)p;
            bfrag[tn].h[1] = *(const bf16x8*)(p + 16);
        }
#pragma unroll
        for (int tm = 0; tm < 2; ++tm)
#pragma unroll
            for (int tn = 0; tn < 4; ++tn)
                acc[tm][tn] = __builtin_amdgcn_wmma_f32_16x16x32_bf16(
                    false, afrag[tm].v, false, bfrag[tn].v,
                    (short)0, acc[tm][tn], false, false);
    };

    // ---- software-pipelined K loop ----
    ldA(0);
    issueB(0, 0);
    stA(0);
    waitB();
    __syncthreads();
#pragma unroll 1
    for (int kt = 0; kt < NKT; ++kt) {
        const int cur = kt & 1;
        if (kt + 1 < NKT) {
            ldA(kt + 1);
            issueB(kt + 1, cur ^ 1);   // safe: barrier at end of kt-1 drained readers
        }
        compute(cur);
        if (kt + 1 < NKT) {
            __syncthreads();
            stA(cur ^ 1);
            waitB();
            __syncthreads();
        }
    }
    __syncthreads();   // LDS about to be reused for the C tile

    // ---- spill C tile to LDS for cross-wave gate fusion ----
#pragma unroll
    for (int tm = 0; tm < 2; ++tm) {
#pragma unroll
        for (int tn = 0; tn < 4; ++tn) {
            const int row0 = wm * 32 + tm * 16 + kh * 8;
            const int col  = wn * 64 + tn * 16 + l15;
#pragma unroll
            for (int r = 0; r < 8; ++r)
                glds[(row0 + r) * NC + col] = acc[tm][tn][r];
        }
    }
    __syncthreads();

    // ---- fused LSTM epilogue ----
    const int hl = tid & 31;
    const int m0 = (tid >> 5) * 16;
    const int hg = hOff + hl;
    const float bi_v = bi[hg], bf_v = bfv[hg], bo_v = bo[hg], bs_v = bs[hg];
#pragma unroll 4
    for (int mm = 0; mm < 16; ++mm) {
        const int m = m0 + mm;
        const float gi = glds[m * NC +      hl] + bi_v;
        const float gf = glds[m * NC + 32 + hl] + bf_v;
        const float go = glds[m * NC + 64 + hl] + bo_v;
        const float gc = glds[m * NC + 96 + hl] + bs_v;
        const float ig = sigmoid_(gi);
        const float fg = sigmoid_(gf);
        const float og = sigmoid_(go);
        const size_t idx = (size_t)(mBase + m) * HD + hg;
        const float ns = fg * old_state[idx] + ig * tanh_(gc);
        outH[idx] = og * tanh_(ns);
        outS[idx] = ns;
    }
}

// ---------------------------------------------------------------------------
// Fallback (no workspace): in-kernel B conversion.
// ---------------------------------------------------------------------------
__global__ __launch_bounds__(256)
void lstm_wmma_fallback(const float* __restrict__ x,
                        const float* __restrict__ pt,
                        const float* __restrict__ pl,
                        const float* __restrict__ old_state,
                        const float* __restrict__ Wi, const float* __restrict__ bi,
                        const float* __restrict__ Wf, const float* __restrict__ bfv,
                        const float* __restrict__ Wo, const float* __restrict__ bo,
                        const float* __restrict__ Ws, const float* __restrict__ bs,
                        float* __restrict__ outH, float* __restrict__ outS)
{
    extern __shared__ char smem_raw[];
    bf16*  const Albase = (bf16*)smem_raw;
    bf16*  const Blbase = (bf16*)(smem_raw + 16384);
    float* const glds   = (float*)smem_raw;

    const int tid  = threadIdx.x;
    const int lane = tid & 31;
    const int wave = tid >> 5;
    const int l15  = lane & 15;
    const int kh   = lane >> 4;
    const int wm   = wave & 3;
    const int wn   = wave >> 2;

    const int mBase = blockIdx.y * MT;
    const int hOff  = blockIdx.x * HT;

    const int ac4 = (tid & 7) * 4;
    const int ar0 = tid >> 3;
    const int bc4  = (tid & 31) * 4;
    const int gate = bc4 >> 5;
    const int gcol = bc4 & 31;
    const int kb   = tid >> 5;
    const float* wsrc = (gate == 0) ? Wi : (gate == 1) ? Wf : (gate == 2) ? Wo : Ws;
    wsrc += (size_t)(hOff + gcol);

    f32x4 areg[4], breg[4];

    auto ldA = [&](int kt) {
        const int k0 = kt * KS;
        const float* src; int stride;
        if (k0 < XRW)           { src = x  + k0;              stride = XRW; }
        else if (k0 < XRW + HD) { src = pt + (k0 - XRW);      stride = HD;  }
        else                    { src = pl + (k0 - XRW - HD); stride = HD;  }
        src += (size_t)(mBase + ar0) * stride + ac4;
#pragma unroll
        for (int j = 0; j < 4; ++j)
            areg[j] = *(const f32x4*)(src + (size_t)(32 * j) * stride);
    };
    auto ldB = [&](int kt) {
        const float* src = wsrc + (size_t)(kt * KS + kb) * HD;
#pragma unroll
        for (int j = 0; j < 4; ++j)
            breg[j] = *(const f32x4*)(src + (size_t)(8 * j) * HD);
    };
    auto stAB = [&](int buf) {
        bf16* Ab = Albase + buf * (MT * KS);
        bf16* Bb = Blbase + buf * (KS * NC);
#pragma unroll
        for (int j = 0; j < 4; ++j) {
            bf16x4 v;
#pragma unroll
            for (int e = 0; e < 4; ++e) v[e] = (bf16)areg[j][e];
            *(bf16x4*)(Ab + (ar0 + 32 * j) * KS + ac4) = v;
        }
#pragma unroll
        for (int j = 0; j < 4; ++j) {
            const int k = kb + 8 * j;
#pragma unroll
            for (int e = 0; e < 4; ++e)
                Bb[(bc4 + e) * KS + k] = (bf16)breg[j][e];
        }
    };

    f32x8 acc[2][4] = {};

    auto compute = [&](int buf) {
        const bf16* Ab = Albase + buf * (MT * KS);
        const bf16* Bb = Blbase + buf * (KS * NC);
        union U16 { bf16x16 v; bf16x8 h[2]; };
        U16 afrag[2], bfrag[4];
#pragma unroll
        for (int tm = 0; tm < 2; ++tm) {
            const bf16* p = Ab + (wm * 32 + tm * 16 + l15) * KS + kh * 8;
            afrag[tm].h[0] = *(const bf16x8*)p;
            afrag[tm].h[1] = *(const bf16x8*)(p + 16);
        }
#pragma unroll
        for (int tn = 0; tn < 4; ++tn) {
            const bf16* p = Bb + (wn * 64 + tn * 16 + l15) * KS + kh * 8;
            bfrag[tn].h[0] = *(const bf16x8*)p;
            bfrag[tn].h[1] = *(const bf16x8*)(p + 16);
        }
#pragma unroll
        for (int tm = 0; tm < 2; ++tm)
#pragma unroll
            for (int tn = 0; tn < 4; ++tn)
                acc[tm][tn] = __builtin_amdgcn_wmma_f32_16x16x32_bf16(
                    false, afrag[tm].v, false, bfrag[tn].v,
                    (short)0, acc[tm][tn], false, false);
    };

    ldA(0); ldB(0);
    stAB(0);
    __syncthreads();
#pragma unroll 1
    for (int kt = 0; kt < NKT; ++kt) {
        const int cur = kt & 1;
        if (kt + 1 < NKT) { ldA(kt + 1); ldB(kt + 1); }
        compute(cur);
        if (kt + 1 < NKT) {
            __syncthreads();
            stAB(cur ^ 1);
            __syncthreads();
        }
    }
    __syncthreads();

#pragma unroll
    for (int tm = 0; tm < 2; ++tm) {
#pragma unroll
        for (int tn = 0; tn < 4; ++tn) {
            const int row0 = wm * 32 + tm * 16 + kh * 8;
            const int col  = wn * 64 + tn * 16 + l15;
#pragma unroll
            for (int r = 0; r < 8; ++r)
                glds[(row0 + r) * NC + col] = acc[tm][tn][r];
        }
    }
    __syncthreads();

    const int hl = tid & 31;
    const int m0 = (tid >> 5) * 16;
    const int hg = hOff + hl;
    const float bi_v = bi[hg], bf_v = bfv[hg], bo_v = bo[hg], bs_v = bs[hg];
#pragma unroll 4
    for (int mm = 0; mm < 16; ++mm) {
        const int m = m0 + mm;
        const float gi = glds[m * NC +      hl] + bi_v;
        const float gf = glds[m * NC + 32 + hl] + bf_v;
        const float go = glds[m * NC + 64 + hl] + bo_v;
        const float gc = glds[m * NC + 96 + hl] + bs_v;
        const float ig = sigmoid_(gi);
        const float fg = sigmoid_(gf);
        const float og = sigmoid_(go);
        const size_t idx = (size_t)(mBase + m) * HD + hg;
        const float ns = fg * old_state[idx] + ig * tanh_(gc);
        outH[idx] = og * tanh_(ns);
        outS[idx] = ns;
    }
}

extern "C" void kernel_launch(void* const* d_in, const int* in_sizes, int n_in,
                              void* d_out, int out_size, void* d_ws, size_t ws_size,
                              hipStream_t stream) {
    (void)in_sizes; (void)n_in; (void)out_size;
    const float* x   = (const float*)d_in[0];
    const float* pt  = (const float*)d_in[1];
    const float* pl  = (const float*)d_in[2];
    const float* os_ = (const float*)d_in[3];
    const float* Wi  = (const float*)d_in[4];
    const float* bi  = (const float*)d_in[5];
    const float* Wf  = (const float*)d_in[6];
    const float* bf_ = (const float*)d_in[7];
    const float* Wo  = (const float*)d_in[8];
    const float* bo  = (const float*)d_in[9];
    const float* Ws  = (const float*)d_in[10];
    const float* bs  = (const float*)d_in[11];
    float* outH = (float*)d_out;
    float* outS = outH + (size_t)BS * HD;

    dim3 grid(HD / HT, BS / MT);   // (8, 128): h-slices fastest so M-tiles hit L2

    if (ws_size >= WS_NEEDED) {
        bf16* wbf = (bf16*)d_ws;
        conv_w_kernel<<<dim3(HD / HT, NKT), 256, 0, stream>>>(Wi, Wf, Wo, Ws, wbf);
        lstm_wmma_async<<<grid, 256, 65536, stream>>>(
            x, pt, pl, os_, wbf, bi, bf_, bo, bs, outH, outS);
    } else {
        lstm_wmma_fallback<<<grid, 256, 65536, stream>>>(
            x, pt, pl, os_, Wi, bi, Wf, bf_, Wo, bo, Ws, bs, outH, outS);
    }
}
